// PVConv_22505628631759
// MI455X (gfx1250) — matrix operand order repaired and checked by reading, not compiled, not run on hardware
//
#include <hip/hip_runtime.h>
#include <hip/hip_bf16.h>

#define BB   4
#define CC   64
#define NN   131072
#define RR   32
#define R3   32768
#define NEG_SLOPE 0.1f
#define EPS3D 1e-4f
#define EPS1D 1e-5f

typedef _Float16 v16h __attribute__((ext_vector_type(16)));
typedef _Float16 v8h  __attribute__((ext_vector_type(8)));
typedef float    v8f  __attribute__((ext_vector_type(8)));
typedef int      vsi4 __attribute__((vector_size(16)));   // matches async-LDS builtin param

#define AS1 __attribute__((address_space(1)))
#define AS3 __attribute__((address_space(3)))

#if __has_builtin(__builtin_amdgcn_global_load_async_to_lds_b128) && \
    __has_builtin(__builtin_amdgcn_s_wait_asynccnt)
#define USE_ASYNC_LDS 1
#else
#define USE_ASYNC_LDS 0
#endif

// Load a 16-half fragment from a 16B-aligned address (LDS or global).
__device__ __forceinline__ v16h load_frag16(const _Float16* p) {
    v8h lo = *(const v8h*)(p);
    v8h hi = *(const v8h*)(p + 8);
    return __builtin_shufflevector(lo, hi, 0,1,2,3,4,5,6,7,8,9,10,11,12,13,14,15);
}

// ---------------- voxelize: per-batch/axis mean (deterministic block reduce) --
__global__ __launch_bounds__(256) void mean_kernel(const float* __restrict__ coords,
                                                   float* __restrict__ mean) {
    int b = blockIdx.x / 3, a = blockIdx.x % 3;
    const float* p = coords + ((size_t)b * 3 + a) * NN;
    float s = 0.f;
    for (int i = threadIdx.x; i < NN; i += 256) s += p[i];
    __shared__ float sm[256];
    sm[threadIdx.x] = s; __syncthreads();
    for (int off = 128; off > 0; off >>= 1) {
        if (threadIdx.x < off) sm[threadIdx.x] += sm[threadIdx.x + off];
        __syncthreads();
    }
    if (threadIdx.x == 0) mean[b * 3 + a] = sm[0] * (1.0f / (float)NN);
}

// ---------------- voxelize: per-batch max ||centered||, scale = 2*max ---------
__global__ __launch_bounds__(256) void scale_kernel(const float* __restrict__ coords,
                                                    const float* __restrict__ mean,
                                                    float* __restrict__ scale) {
    int b = blockIdx.x;
    float mx = mean[b*3+0], my = mean[b*3+1], mz = mean[b*3+2];
    const float* p = coords + (size_t)b * 3 * NN;
    float m = 0.f;
    for (int i = threadIdx.x; i < NN; i += 256) {
        float x = p[i] - mx, y = p[i + NN] - my, z = p[i + 2*NN] - mz;
        m = fmaxf(m, sqrtf(x*x + y*y + z*z));
    }
    __shared__ float sm[256];
    sm[threadIdx.x] = m; __syncthreads();
    for (int off = 128; off > 0; off >>= 1) {
        if (threadIdx.x < off) sm[threadIdx.x] = fmaxf(sm[threadIdx.x], sm[threadIdx.x + off]);
        __syncthreads();
    }
    if (threadIdx.x == 0) scale[b] = 2.0f * sm[0];
}

// ---------------- voxelize: scatter-add features into [b][vox][c] grid -------
__global__ __launch_bounds__(256) void scatter_kernel(const float* __restrict__ feat,
                                                      const float* __restrict__ coords,
                                                      const float* __restrict__ mean,
                                                      const float* __restrict__ scale,
                                                      float* __restrict__ nc,
                                                      float* __restrict__ cnt,
                                                      float* __restrict__ gacc) {
    int gid = blockIdx.x * 256 + threadIdx.x;   // b*NN + n
    int b = gid >> 17, n = gid & (NN - 1);
    const float* cb = coords + (size_t)b * 3 * NN;
    float sc = scale[b];
    float v[3];
#pragma unroll
    for (int a = 0; a < 3; ++a) {
        float cen = cb[(size_t)a * NN + n] - mean[b*3+a];
        float t = (cen / sc + 0.5f) * (float)RR;
        t = fminf(fmaxf(t, 0.f), (float)(RR - 1));
        v[a] = t;
        nc[((size_t)b * 3 + a) * NN + n] = t;
    }
    int ix = (int)rintf(v[0]), iy = (int)rintf(v[1]), iz = (int)rintf(v[2]);
    int id = (ix * RR + iy) * RR + iz;
    atomicAdd(&cnt[(size_t)b * R3 + id], 1.0f);
    const float* fb = feat + (size_t)b * CC * NN + n;
    float* gb = gacc + ((size_t)b * R3 + id) * CC;
    for (int c = 0; c < CC; ++c)
        atomicAdd(&gb[c], fb[(size_t)c * NN]);
}

// ---------------- average + f32 -> f16 grid ([b][vox][c] layout) -------------
__global__ __launch_bounds__(256) void avg_kernel(const float* __restrict__ gacc,
                                                  const float* __restrict__ cnt,
                                                  _Float16* __restrict__ g16) {
    size_t i = (size_t)blockIdx.x * 256 + threadIdx.x;   // B*R3*CC total
    size_t b = i >> 21;                                   // R3*CC = 2^21
    size_t vox = (i >> 6) & (R3 - 1);
    float c = fmaxf(cnt[b * R3 + vox], 1.0f);
    g16[i] = (_Float16)(gacc[i] / c);
}

// ---------------- prepack conv weights into CDNA5 A-fragment layout ----------
// frag f = ((tap27*2)+kc)*4 + mtile ; elem i of lane L:
//   o = mtile*16 + (L&15) ; c = kc*32 + (L>>4)*16 + i
__global__ __launch_bounds__(256) void prepack_conv_kernel(const float* __restrict__ w,
                                                           _Float16* __restrict__ dst) {
    int id = blockIdx.x * 256 + threadIdx.x;
    if (id >= 216 * 32) return;
    int lane = id & 31, f = id >> 5;
    int mtile = f & 3, kc = (f >> 2) & 1, tap = f >> 3;     // tap in 0..26
    int o = mtile * 16 + (lane & 15);
    _Float16* d = dst + (size_t)f * 512 + lane * 16;
#pragma unroll
    for (int i = 0; i < 16; ++i) {
        int c = kc * 32 + ((lane >> 4) << 4) + i;
        d[i] = (_Float16)w[(size_t)o * (CC * 27) + c * 27 + tap];
    }
}

__global__ __launch_bounds__(256) void prepack_mlp_kernel(const float* __restrict__ w,
                                                          _Float16* __restrict__ dst) {
    int id = blockIdx.x * 256 + threadIdx.x;
    if (id >= 8 * 32) return;
    int lane = id & 31, f = id >> 5;
    int mtile = f & 3, kc = (f >> 2) & 1;
    int o = mtile * 16 + (lane & 15);
    _Float16* d = dst + (size_t)f * 512 + lane * 16;
#pragma unroll
    for (int i = 0; i < 16; ++i) {
        int c = kc * 32 + ((lane >> 4) << 4) + i;
        d[i] = (_Float16)w[(size_t)o * CC + c];
    }
}

// ---------------- 3x3x3 conv as implicit GEMM via WMMA + fused BN/LeakyReLU --
// Grids are channel-innermost: g[b][x][y][z][c]. block = (b,x,y), 128 threads
// = 4 waves; wave w owns output channels [16w,16w+16); N = 32 z as 2 frags.
template <bool OUTF16>
__global__ __launch_bounds__(128) void conv_kernel(const _Float16* __restrict__ gin,
                                                   const _Float16* __restrict__ wpack,
                                                   const float* __restrict__ bng,
                                                   const float* __restrict__ bnb,
                                                   const float* __restrict__ bnm,
                                                   const float* __restrict__ bnv,
                                                   _Float16* __restrict__ out16,
                                                   float* __restrict__ out32) {
    __shared__ _Float16 in_s[9][34][72];   // [dxdy][z+1][cin], padded row
    __shared__ float s_sc[64], s_bi[64];
    int blk = blockIdx.x;
    int b = blk >> 10, xy = blk & 1023, x = xy >> 5, y = xy & 31;
    int tid = threadIdx.x;

    if (tid < 64) {
        float s = bng[tid] * rsqrtf(bnv[tid] + EPS3D);
        s_sc[tid] = s;
        s_bi[tid] = bnb[tid] - bnm[tid] * s;
    }
    // stage halo: each halo row (t,zz) = 64 contiguous halves = 8 x b128 chunks
    for (int i = tid; i < 9 * 34 * 8; i += 128) {
        int row = i >> 3, ch = i & 7;
        int t = row / 34, zz = row - t * 34;
        int X = x + t / 3 - 1, Y = y + t % 3 - 1, Z = zz - 1;
        _Float16* ldst = &in_s[t][zz][ch * 8];
        if ((unsigned)X < 32u && (unsigned)Y < 32u && (unsigned)Z < 32u) {
            const _Float16* src =
                gin + ((((size_t)b * 1024 + X * 32 + Y) * 32 + Z) * CC + ch * 8);
#if USE_ASYNC_LDS
            __builtin_amdgcn_global_load_async_to_lds_b128(
                (AS1 vsi4*)(AS1 void*)(void*)const_cast<_Float16*>(src),
                (AS3 vsi4*)(AS3 void*)(void*)ldst, 0, 0);
#else
            *(v8h*)ldst = *(const v8h*)src;
#endif
        } else {
            v8h z8 = {};
            *(v8h*)ldst = z8;
        }
    }
#if USE_ASYNC_LDS
    __builtin_amdgcn_s_wait_asynccnt(0);
#endif
    __syncthreads();

    int wave = tid >> 5, lane = tid & 31, lm = lane & 15, lh = lane >> 4;
    v8f acc0 = {}, acc1 = {};
    for (int t = 0; t < 9; ++t) {
        for (int dz = 0; dz < 3; ++dz) {
            int zz0 = lm + dz;            // input-z index (+1 bias) for ntile 0
#pragma unroll
            for (int kc = 0; kc < 2; ++kc) {
                int f = ((t * 3 + dz) * 2 + kc) * 4 + wave;
                v16h a  = load_frag16(wpack + (size_t)f * 512 + lane * 16);
                v16h b0 = load_frag16(&in_s[t][zz0][kc * 32 + lh * 16]);
                v16h b1 = load_frag16(&in_s[t][zz0 + 16][kc * 32 + lh * 16]);
                acc0 = __builtin_amdgcn_wmma_f32_16x16x32_f16(false, a, false, b0,
                                                              (short)0, acc0, false, false);
                acc1 = __builtin_amdgcn_wmma_f32_16x16x32_f16(false, a, false, b1,
                                                              (short)0, acc1, false, false);
            }
        }
    }
    // epilogue: 8 consecutive channels per lane -> vector stores
    int och = wave * 16 + (lh << 3);
    size_t rowbase = ((size_t)b * 1024 + x * 32 + y) * 32;
    if (OUTF16) {
        v8h o0, o1;
#pragma unroll
        for (int j = 0; j < 8; ++j) {
            int o = och + j;
            float s = s_sc[o], bi = s_bi[o];
            float f0 = acc0[j] * s + bi; f0 = f0 >= 0.f ? f0 : NEG_SLOPE * f0;
            float f1 = acc1[j] * s + bi; f1 = f1 >= 0.f ? f1 : NEG_SLOPE * f1;
            o0[j] = (_Float16)f0; o1[j] = (_Float16)f1;
        }
        *(v8h*)&out16[(rowbase + lm) * CC + och]      = o0;
        *(v8h*)&out16[(rowbase + lm + 16) * CC + och] = o1;
    } else {
        v8f o0, o1;
#pragma unroll
        for (int j = 0; j < 8; ++j) {
            int o = och + j;
            float s = s_sc[o], bi = s_bi[o];
            float f0 = acc0[j] * s + bi; f0 = f0 >= 0.f ? f0 : NEG_SLOPE * f0;
            float f1 = acc1[j] * s + bi; f1 = f1 >= 0.f ? f1 : NEG_SLOPE * f1;
            o0[j] = f0; o1[j] = f1;
        }
        *(v8f*)&out32[(rowbase + lm) * CC + och]      = o0;
        *(v8f*)&out32[(rowbase + lm + 16) * CC + och] = o1;
    }
}

// ---------------- fused pointwise-MLP GEMM + BN/ReLU + trilinear devox + add -
// grid2 is channel-innermost: grid2[b][vox][c]
__global__ __launch_bounds__(128) void mlpdevox_kernel(const float* __restrict__ feat,
                                                       const _Float16* __restrict__ wpackm,
                                                       const float* __restrict__ mlp_b,
                                                       const float* __restrict__ bng,
                                                       const float* __restrict__ bnb,
                                                       const float* __restrict__ bnm,
                                                       const float* __restrict__ bnv,
                                                       const float* __restrict__ nc,
                                                       const float* __restrict__ grid2,
                                                       float* __restrict__ out) {
    __shared__ _Float16 fs[64][72];        // [point][cin], padded
    __shared__ float s_sc[64], s_bi[64];
    int tid = threadIdx.x, blk = blockIdx.x;
    int b = blk >> 11;                      // NN/64 = 2048 blocks per batch
    int n0 = (blk & 2047) << 6;

    if (tid < 64) {
        float s = bng[tid] * rsqrtf(bnv[tid] + EPS1D);
        s_sc[tid] = s;
        s_bi[tid] = bnb[tid] + (mlp_b[tid] - bnm[tid]) * s;
    }
    // stage 64 points x 64 channels with float4 loads (n is contiguous)
    for (int i = tid; i < 1024; i += 128) {
        int c = i >> 4, p4 = (i & 15) << 2;
        float4 v = *(const float4*)&feat[((size_t)b * CC + c) * NN + n0 + p4];
        fs[p4 + 0][c] = (_Float16)v.x;
        fs[p4 + 1][c] = (_Float16)v.y;
        fs[p4 + 2][c] = (_Float16)v.z;
        fs[p4 + 3][c] = (_Float16)v.w;
    }
    __syncthreads();

    int wave = tid >> 5, lane = tid & 31, lm = lane & 15, lh = lane >> 4;
    v8f acc[4] = {{}, {}, {}, {}};
#pragma unroll
    for (int kc = 0; kc < 2; ++kc) {
        v16h bf = load_frag16(&fs[wave * 16 + lm][kc * 32 + lh * 16]);
#pragma unroll
        for (int mt = 0; mt < 4; ++mt) {
            v16h a = load_frag16(wpackm + (size_t)(kc * 4 + mt) * 512 + lane * 16);
            acc[mt] = __builtin_amdgcn_wmma_f32_16x16x32_f16(false, a, false, bf,
                                                             (short)0, acc[mt], false, false);
        }
    }
    // trilinear setup (per lane's point)
    int n = n0 + wave * 16 + lm;
    float ncx = nc[((size_t)b * 3 + 0) * NN + n];
    float ncy = nc[((size_t)b * 3 + 1) * NN + n];
    float ncz = nc[((size_t)b * 3 + 2) * NN + n];
    int lx = (int)floorf(ncx), ly = (int)floorf(ncy), lz = (int)floorf(ncz);
    float fx = ncx - (float)lx, fy = ncy - (float)ly, fz = ncz - (float)lz;
    int hx = min(lx + 1, RR - 1), hy = min(ly + 1, RR - 1), hz = min(lz + 1, RR - 1);
    int idx8[8]; float w8[8];
    int k = 0;
#pragma unroll
    for (int dx = 0; dx < 2; ++dx)
#pragma unroll
        for (int dy = 0; dy < 2; ++dy)
#pragma unroll
            for (int dz = 0; dz < 2; ++dz) {
                int X = dx ? hx : lx, Y = dy ? hy : ly, Z = dz ? hz : lz;
                idx8[k] = (X * RR + Y) * RR + Z;
                w8[k] = (dx ? fx : 1.f - fx) * (dy ? fy : 1.f - fy) * (dz ? fz : 1.f - fz);
                ++k;
            }
    const float* gb = grid2 + (size_t)b * R3 * CC;
#pragma unroll
    for (int mt = 0; mt < 4; ++mt) {
        int obase = mt * 16 + (lh << 3);
        v8f vf = {};
#pragma unroll
        for (int q = 0; q < 8; ++q) {
            v8f g = *(const v8f*)&gb[(size_t)idx8[q] * CC + obase];
            vf += g * w8[q];
        }
#pragma unroll
        for (int j = 0; j < 8; ++j) {
            int o = obase + j;
            float pf = fmaxf(acc[mt][j] * s_sc[o] + s_bi[o], 0.f);
            out[((size_t)b * CC + o) * NN + n] = pf + vf[j];
        }
    }
}

extern "C" void kernel_launch(void* const* d_in, const int* in_sizes, int n_in,
                              void* d_out, int out_size, void* d_ws, size_t ws_size,
                              hipStream_t stream) {
    const float* features = (const float*)d_in[0];
    const float* coords   = (const float*)d_in[1];
    const float* conv1_w  = (const float*)d_in[2];
    const float* bn1_g = (const float*)d_in[3];
    const float* bn1_b = (const float*)d_in[4];
    const float* bn1_m = (const float*)d_in[5];
    const float* bn1_v = (const float*)d_in[6];
    const float* conv2_w  = (const float*)d_in[7];
    const float* bn2_g = (const float*)d_in[8];
    const float* bn2_b = (const float*)d_in[9];
    const float* bn2_m = (const float*)d_in[10];
    const float* bn2_v = (const float*)d_in[11];
    const float* mlp_w = (const float*)d_in[12];
    const float* mlp_b = (const float*)d_in[13];
    const float* bn3_g = (const float*)d_in[14];
    const float* bn3_b = (const float*)d_in[15];
    const float* bn3_m = (const float*)d_in[16];
    const float* bn3_v = (const float*)d_in[17];
    float* out = (float*)d_out;

    char* ws = (char*)d_ws;
    size_t off = 0;
    auto alloc = [&](size_t bytes) -> void* {
        void* p = ws + off;
        off += (bytes + 255) & ~(size_t)255;
        return p;
    };
    const size_t gridElems = (size_t)BB * CC * R3;            // 8,388,608
    float*    gacc   = (float*)alloc(gridElems * 4);          // reused as grid2 (f32)
    _Float16* grid0h = (_Float16*)alloc(gridElems * 2);
    _Float16* grid1h = (_Float16*)alloc(gridElems * 2);
    float*    ncbuf  = (float*)alloc((size_t)BB * 3 * NN * 4);
    float*    cnt    = (float*)alloc((size_t)BB * R3 * 4);
    float*    meanb  = (float*)alloc(64);
    float*    scaleb = (float*)alloc(64);
    _Float16* wpack1 = (_Float16*)alloc((size_t)216 * 512 * 2);
    _Float16* wpack2 = (_Float16*)alloc((size_t)216 * 512 * 2);
    _Float16* wpackm = (_Float16*)alloc((size_t)8 * 512 * 2);
    float*    grid2  = gacc;   // conv2 output reuses scatter accumulator

    (void)hipMemsetAsync(gacc, 0, gridElems * 4, stream);
    (void)hipMemsetAsync(cnt, 0, (size_t)BB * R3 * 4, stream);

    // weight prepack (independent of data pipeline)
    prepack_conv_kernel<<<27, 256, 0, stream>>>(conv1_w, wpack1);
    prepack_conv_kernel<<<27, 256, 0, stream>>>(conv2_w, wpack2);
    prepack_mlp_kernel<<<1, 256, 0, stream>>>(mlp_w, wpackm);

    // voxelize
    mean_kernel<<<BB * 3, 256, 0, stream>>>(coords, meanb);
    scale_kernel<<<BB, 256, 0, stream>>>(coords, meanb, scaleb);
    scatter_kernel<<<(BB * NN) / 256, 256, 0, stream>>>(features, coords, meanb, scaleb,
                                                        ncbuf, cnt, gacc);
    avg_kernel<<<(int)(gridElems / 256), 256, 0, stream>>>(gacc, cnt, grid0h);

    // conv1 (f16 out) -> conv2 (f32 out), BN+LeakyReLU fused
    conv_kernel<true><<<BB * 1024, 128, 0, stream>>>(grid0h, wpack1, bn1_g, bn1_b, bn1_m,
                                                     bn1_v, grid1h, nullptr);
    conv_kernel<false><<<BB * 1024, 128, 0, stream>>>(grid1h, wpack2, bn2_g, bn2_b, bn2_m,
                                                      bn2_v, nullptr, grid2);

    // fused pointwise MLP + BN/ReLU + trilinear devoxelize + add
    mlpdevox_kernel<<<BB * (NN / 64), 128, 0, stream>>>(features, wpackm, mlp_b, bn3_g,
                                                        bn3_b, bn3_m, bn3_v, ncbuf, grid2,
                                                        out);

    // second tuple output: coords passthrough
    (void)hipMemcpyAsync(out + (size_t)BB * CC * NN, coords, (size_t)BB * 3 * NN * 4,
                         hipMemcpyDeviceToDevice, stream);
}
